// OuterProductMean_43044162241196
// MI455X (gfx1250) — matrix operand order repaired and checked by reading, not compiled, not run on hardware
//
#include <hip/hip_runtime.h>
#include <hip/hip_bf16.h>

typedef __attribute__((ext_vector_type(16))) __bf16 v16bf;
typedef __attribute__((ext_vector_type(8)))  __bf16 v8bf;
typedef __attribute__((ext_vector_type(8)))  float  v8f;
typedef __attribute__((ext_vector_type(4)))  float  v4f;

#define S_DIM 384
#define B_DIM 256
#define CM    256
#define CH    32
#define CZ    128

// ---------------------------------------------------------------------------
// CDNA5 async global->LDS DMA (ASYNCcnt path), per ISA 15.18.3 / 08_async.
// ---------------------------------------------------------------------------
__device__ __forceinline__ void async_ld16(unsigned lds_byte, const void* gptr) {
    asm volatile("global_load_async_to_lds_b128 %0, %1, off"
                 :: "v"(lds_byte), "v"((unsigned long long)(uintptr_t)gptr)
                 : "memory");
}
__device__ __forceinline__ void wait_async0() {
    asm volatile("s_wait_asynccnt 0x0" ::: "memory");
}
__device__ __forceinline__ unsigned lds_off(const void* p) {
    return (unsigned)(uintptr_t)p;     // LDS generic ptr: low 32 bits = LDS addr
}

// ---------------------------------------------------------------------------
// Kernel 0: Wot[e][cd] = (bf16) Wo[cd][e]   (128 x 1024, K-contiguous rows)
// ---------------------------------------------------------------------------
__global__ void k_wot(const float* __restrict__ Wo, unsigned short* __restrict__ Wot_u) {
    __bf16* Wot = (__bf16*)Wot_u;
    int idx = blockIdx.x * 256 + threadIdx.x;      // 131072 total
    int e  = idx >> 10;
    int cd = idx & 1023;
    Wot[idx] = (__bf16)Wo[cd * CZ + e];
}

// ---------------------------------------------------------------------------
// Kernel 1 (WMMA): q,k projections as a 256x64x256 GEMM per s.
//   Block = one s, 8 waves; wave w owns b-tiles {2w, 2w+1}; N = [Wq|Wk].
//   A-frags built from f32 m on the fly (x g_in, cvt bf16); row sum(m^2)
//   accumulated during frag build (f32) -> rsqrt applied to D + bias.
//   Output: Qt[s][c][b], Kt[s][c][b] bf16 (b innermost, WMMA K-contiguous).
// ---------------------------------------------------------------------------
#define WB_STR 264        // padded LDS row stride (halves) for W^T[n][x]

__global__ __launch_bounds__(256) void k_proj(
    const float* __restrict__ m,  const float* __restrict__ g_in,
    const float* __restrict__ Wq, const float* __restrict__ bq,
    const float* __restrict__ Wk, const float* __restrict__ bk,
    unsigned short* __restrict__ Qt_u, unsigned short* __restrict__ Kt_u)
{
    __bf16* Qt = (__bf16*)Qt_u;
    __bf16* Kt = (__bf16*)Kt_u;

    __shared__ __align__(16) __bf16 sW[64 * WB_STR];   // [n][x] bf16, 33.8 KB
    __shared__ __align__(16) float  sG[CM];            // g_in

    const int s    = blockIdx.x;
    const int tid  = threadIdx.x;
    const int w    = tid >> 5;
    const int lane = tid & 31;
    const int m16  = lane & 15, hi = lane >> 4;

    sG[tid] = g_in[tid];
    {   // stage [Wq|Wk]^T as bf16 rows [n][x]
        int n = tid >> 2, xs = tid & 3;
        const float* col = (n < CH) ? (Wq + n) : (Wk + (n - CH));
        for (int x = xs * 64; x < xs * 64 + 64; ++x)
            sW[n * WB_STR + x] = (__bf16)col[x * CH];
    }
    __syncthreads();

    v8f acc[2][4];                    // [mt][nt]
#pragma unroll
    for (int mt = 0; mt < 2; ++mt)
#pragma unroll
        for (int nt = 0; nt < 4; ++nt)
#pragma unroll
            for (int r = 0; r < 8; ++r) acc[mt][nt][r] = 0.f;
    float ssq[2] = {0.f, 0.f};

    for (int k = 0; k < 8; ++k) {     // K = 256 / 32
        const float* pg = sG + k * 32 + hi * 8;
        v4f g[4];
        g[0] = *(const v4f*)pg;       g[1] = *(const v4f*)(pg + 4);
        g[2] = *(const v4f*)(pg + 16); g[3] = *(const v4f*)(pg + 20);

        v16bf afr[2];
#pragma unroll
        for (int mt = 0; mt < 2; ++mt) {
            int b = (w * 2 + mt) * 16 + m16;
            const float* px = m + ((size_t)b * S_DIM + s) * CM + k * 32 + hi * 8;
            v4f f[4];
            f[0] = *(const v4f*)px;        f[1] = *(const v4f*)(px + 4);
            f[2] = *(const v4f*)(px + 16); f[3] = *(const v4f*)(px + 20);
#pragma unroll
            for (int q = 0; q < 4; ++q)
#pragma unroll
                for (int e = 0; e < 4; ++e) {
                    float x = f[q][e];
                    ssq[mt] = fmaf(x, x, ssq[mt]);            // raw m^2 (f32)
                    afr[mt][q * 4 + e] = (__bf16)(x * g[q][e]);
                }
        }
#pragma unroll
        for (int nt = 0; nt < 4; ++nt) {
            const __bf16* p = sW + (nt * 16 + m16) * WB_STR + k * 32 + hi * 8;
            v8bf lo = *(const v8bf*)p, hp = *(const v8bf*)(p + 16);
            v16bf bfr;
#pragma unroll
            for (int t = 0; t < 8; ++t) { bfr[t] = lo[t]; bfr[t + 8] = hp[t]; }
            acc[0][nt] = __builtin_amdgcn_wmma_f32_16x16x32_bf16(
                false, afr[0], false, bfr, (short)0, acc[0][nt], false, false);
            acc[1][nt] = __builtin_amdgcn_wmma_f32_16x16x32_bf16(
                false, afr[1], false, bfr, (short)0, acc[1][nt], false, false);
        }
    }

    // close row reductions: lanes (m16, m16+16) each hold half of row m16
    float rsr[2][8];
#pragma unroll
    for (int mt = 0; mt < 2; ++mt) {
        float sq = ssq[mt] + __shfl_xor(ssq[mt], 16);
        float rs = rsqrtf(sq * (1.f / CM) + 1e-5f);           // row = m16 of tile
#pragma unroll
        for (int r = 0; r < 8; ++r)
            rsr[mt][r] = __shfl(rs, r + 8 * hi);              // row of D VGPR r
    }
#pragma unroll
    for (int mt = 0; mt < 2; ++mt)
#pragma unroll
        for (int nt = 0; nt < 4; ++nt) {
            int n = nt * 16 + m16;
            float bias = (n < CH) ? bq[n] : bk[n - CH];
            __bf16* dst = (n < CH) ? (Qt + (s * CH + n) * B_DIM)
                                   : (Kt + (s * CH + (n - CH)) * B_DIM);
#pragma unroll
            for (int r = 0; r < 8; ++r) {
                int b = (w * 2 + mt) * 16 + r + 8 * hi;
                dst[b] = (__bf16)(acc[mt][nt][r] * rsr[mt][r] + bias);
            }
        }
}

// ---------------------------------------------------------------------------
// Kernel 2: fused outer-product (over b) + Wo contraction + rms_norm.
// Block = (2 i's, 16 j's) = 32 pairs, 16 waves.
//  Phase A: wave w owns j = j0+w for both i's. 8 b-chunks of 32, staged by
//           async-to-LDS DMA, double buffered; 8 WMMAs/wave/chunk.
//  Phase B: outer -> bf16 A2[32][1024] in LDS; 16 waves split-K GEMM vs
//           Wot (B-frag reused across both M-tiles); LDS f32 reduction;
//           bias + rms_norm + coalesced store.
// ---------------------------------------------------------------------------
#define SK_STR   40       // padded row stride (halves) for [row][32 b] chunks
#define A2_STR   1032     // padded row stride (halves) for A2[pair][1024]
#define OFF_SK0  0        // 512*40*2 = 40960 B   (K chunk, buffer 0)
#define OFF_SK1  40960    // 40960 B              (K chunk, buffer 1)
#define OFF_SQ0  81920    // 64*40*2 = 5120 B     (Q chunk, buffer 0)
#define OFF_SQ1  87040    // 5120 B               (Q chunk, buffer 1)
#define OFF_RED  92160    // 8*2*256*4 = 16384 B  (phase-B split-K partials)
#define OFF_SSQ  108544   // 32*4 = 128 B
#define SMEM_SZ  108672

__global__ __launch_bounds__(512) void k_fused(
    const unsigned short* __restrict__ Qt_u, const unsigned short* __restrict__ Kt_u,
    const unsigned short* __restrict__ Wot_u,
    const float* __restrict__ bo, const float* __restrict__ g_out,
    float* __restrict__ out)
{
    const __bf16* Qt  = (const __bf16*)Qt_u;
    const __bf16* Kt  = (const __bf16*)Kt_u;
    const __bf16* Wot = (const __bf16*)Wot_u;

    extern __shared__ unsigned char smem[];
    float* sSsq = (float*)(smem + OFF_SSQ);
    float* sRed = (float*)(smem + OFF_RED);

    const int tid  = threadIdx.x;
    const int w    = tid >> 5;        // wave 0..15 == local j index (phase A)
    const int lane = tid & 31;
    const int m16  = lane & 15;
    const int hi   = lane >> 4;
    const int i0   = blockIdx.y * 2;
    const int j0   = blockIdx.x * 16;

    if (tid < 32) sSsq[tid] = 0.f;
    __builtin_prefetch(Wot + (size_t)(w & 7) * 16 * 1024, 0, 1);

    v8f acc[2][2][2];                 // [ii][ch][dh]
#pragma unroll
    for (int ii = 0; ii < 2; ++ii)
#pragma unroll
        for (int ch = 0; ch < 2; ++ch)
#pragma unroll
            for (int dh = 0; dh < 2; ++dh)
#pragma unroll
                for (int r = 0; r < 8; ++r) acc[ii][ch][dh][r] = 0.f;

    // per-thread staging coordinates
    const int kjj = tid >> 5, kd = tid & 31;                 // K row per thread
    const __bf16* ksrc0 = Kt + ((j0 + kjj) * CH + kd) * B_DIM;
    const unsigned kdst0 = lds_off(smem) + (unsigned)((kjj * 32 + kd) * SK_STR * 2);
    const int qr = tid >> 2, qseg = tid & 3;                 // Q row (tid<256)
    const __bf16* qsrc0 = Qt + ((i0 + (qr >> 5)) * CH + (qr & 31)) * B_DIM + qseg * 8;
    const unsigned qdst0 = lds_off(smem) + (unsigned)(qr * SK_STR * 2 + qseg * 16);

    // ---- stage chunk 0 into buffer 0 (async DMA) ----
    {
#pragma unroll
        for (int q8 = 0; q8 < 4; ++q8)
            async_ld16(kdst0 + OFF_SK0 + q8 * 16, ksrc0 + q8 * 8);
        if (tid < 256) async_ld16(qdst0 + OFF_SQ0, qsrc0);
    }
    wait_async0();
    __syncthreads();

    // ---------------- Phase A ----------------
    for (int bc = 0; bc < 8; ++bc) {
        const int cur = bc & 1;
        if (bc < 7) {                 // prefetch next chunk into other buffer
            const int nxt = bc + 1;
            const unsigned skoff = (cur ? OFF_SK0 : OFF_SK1);
            const unsigned sqoff = (cur ? OFF_SQ0 : OFF_SQ1);
#pragma unroll
            for (int q8 = 0; q8 < 4; ++q8)
                async_ld16(kdst0 + skoff + q8 * 16, ksrc0 + nxt * 32 + q8 * 8);
            if (tid < 256) async_ld16(qdst0 + sqoff, qsrc0 + nxt * 32);
        }
        const __bf16* sKc = (const __bf16*)(smem + (cur ? OFF_SK1 : OFF_SK0));
        const __bf16* sQc = (const __bf16*)(smem + (cur ? OFF_SQ1 : OFF_SQ0));

        v16bf bfr[2];
#pragma unroll
        for (int dh = 0; dh < 2; ++dh) {   // B frag: N=d, per-lane K {hi*8.., 16+hi*8..}
            const __bf16* p = sKc + (w * 32 + dh * 16 + m16) * SK_STR + hi * 8;
            v8bf lo = *(const v8bf*)p, hp = *(const v8bf*)(p + 16);
#pragma unroll
            for (int t = 0; t < 8; ++t) { bfr[dh][t] = lo[t]; bfr[dh][t + 8] = hp[t]; }
        }
#pragma unroll
        for (int ii = 0; ii < 2; ++ii)
#pragma unroll
            for (int ch = 0; ch < 2; ++ch) {   // A frag: K {hi*8.., 16+hi*8..}
                const __bf16* p = sQc + (ii * 32 + ch * 16 + m16) * SK_STR + hi * 8;
                v8bf lo = *(const v8bf*)p, hp = *(const v8bf*)(p + 16);
                v16bf a;
#pragma unroll
                for (int t = 0; t < 8; ++t) { a[t] = lo[t]; a[t + 8] = hp[t]; }
                acc[ii][ch][0] = __builtin_amdgcn_wmma_f32_16x16x32_bf16(
                    false, a, false, bfr[0], (short)0, acc[ii][ch][0], false, false);
                acc[ii][ch][1] = __builtin_amdgcn_wmma_f32_16x16x32_bf16(
                    false, a, false, bfr[1], (short)0, acc[ii][ch][1], false, false);
            }
        wait_async0();                // my DMA done; barrier orders everyone
        __syncthreads();
    }

    // ------------- Phase A->B: outer -> bf16 A2[ii*16+w][c*32+d] ------------
    __bf16* A2 = (__bf16*)smem;       // overlay (32*1032*2 = 66048 B)
#pragma unroll
    for (int ii = 0; ii < 2; ++ii)
#pragma unroll
        for (int ch = 0; ch < 2; ++ch)
#pragma unroll
            for (int dh = 0; dh < 2; ++dh)
#pragma unroll
                for (int r = 0; r < 8; ++r) {
                    int c = ch * 16 + hi * 8 + r;      // C/D: M = r + 8*hi
                    int d = dh * 16 + m16;             //      N = lane%16
                    A2[(ii * 16 + w) * A2_STR + c * 32 + d] = (__bf16)acc[ii][ch][dh][r];
                }
    __syncthreads();

    // ------- Phase B: z = A2 @ Wot + bo, rms_norm.  16 waves, split-K -------
    const int kh = w >> 3, eT = w & 7;
    const int eo = eT * 16 + m16;                      // e channel
    v8f c2[2];
#pragma unroll
    for (int mt = 0; mt < 2; ++mt)
#pragma unroll
        for (int r = 0; r < 8; ++r) c2[mt][r] = 0.f;

    for (int s2 = kh * 16; s2 < kh * 16 + 16; ++s2) {  // K half per wave group
        const __bf16* pb = Wot + eo * 1024 + s2 * 32 + hi * 8;
        v8bf blo = *(const v8bf*)pb, bhp = *(const v8bf*)(pb + 16);
        v16bf b;
#pragma unroll
        for (int t = 0; t < 8; ++t) { b[t] = blo[t]; b[t + 8] = bhp[t]; }
#pragma unroll
        for (int mt = 0; mt < 2; ++mt) {               // reuse B-frag, 2 M-tiles
            const __bf16* pa = A2 + (mt * 16 + m16) * A2_STR + s2 * 32 + hi * 8;
            v8bf alo = *(const v8bf*)pa, ahp = *(const v8bf*)(pa + 16);
            v16bf a;
#pragma unroll
            for (int t = 0; t < 8; ++t) { a[t] = alo[t]; a[t + 8] = ahp[t]; }
            c2[mt] = __builtin_amdgcn_wmma_f32_16x16x32_bf16(
                false, a, false, b, (short)0, c2[mt], false, false);
        }
    }
    if (kh == 1) {                                     // dump K-half partials
#pragma unroll
        for (int mt = 0; mt < 2; ++mt)
            *(v8f*)(sRed + (eT * 2 + mt) * 256 + lane * 8) = c2[mt];
    }
    __syncthreads();

    float zr[2][8];
    if (kh == 0) {
        float bov = bo[eo];
#pragma unroll
        for (int mt = 0; mt < 2; ++mt) {
            v8f oth = *(const v8f*)(sRed + (eT * 2 + mt) * 256 + lane * 8);
#pragma unroll
            for (int r = 0; r < 8; ++r) {
                float z = c2[mt][r] + oth[r] + bov;
                zr[mt][r] = z;
                float p = z * z;                       // reduce 16 e's in group
                p += __shfl_xor(p, 1);
                p += __shfl_xor(p, 2);
                p += __shfl_xor(p, 4);
                p += __shfl_xor(p, 8);
                if (m16 == 0) atomicAdd(&sSsq[mt * 16 + hi * 8 + r], p);
            }
        }
    }
    __syncthreads();
    if (kh == 0) {
        float gov = g_out[eo];
#pragma unroll
        for (int mt = 0; mt < 2; ++mt)
#pragma unroll
            for (int r = 0; r < 8; ++r) {
                int jj = hi * 8 + r;                   // local j of this row
                float rs = rsqrtf(sSsq[mt * 16 + jj] * (1.f / CZ) + 1e-5f);
                out[((size_t)(i0 + mt) * S_DIM + (j0 + jj)) * CZ + eo] =
                    zr[mt][r] * rs * gov;
            }
    }
}

// ---------------------------------------------------------------------------
extern "C" void kernel_launch(void* const* d_in, const int* in_sizes, int n_in,
                              void* d_out, int out_size, void* d_ws, size_t ws_size,
                              hipStream_t stream) {
    const float* m    = (const float*)d_in[0];
    const float* g_in = (const float*)d_in[1];
    const float* Wq   = (const float*)d_in[2];
    const float* bq   = (const float*)d_in[3];
    const float* Wk   = (const float*)d_in[4];
    const float* bk   = (const float*)d_in[5];
    const float* Wo   = (const float*)d_in[6];
    const float* bo   = (const float*)d_in[7];
    const float* gout = (const float*)d_in[8];
    float* out = (float*)d_out;

    const size_t QK_ELEMS = (size_t)S_DIM * CH * B_DIM;     // 3,145,728
    unsigned short* Qt  = (unsigned short*)d_ws;
    unsigned short* Kt  = Qt + QK_ELEMS;
    unsigned short* Wot = Kt + QK_ELEMS;                    // ~12.8 MB total

    k_wot <<<512, 256, 0, stream>>>(Wo, Wot);
    k_proj<<<S_DIM, 256, 0, stream>>>(m, g_in, Wq, bq, Wk, bk, Qt, Kt);
    k_fused<<<dim3(S_DIM / 16, S_DIM / 2), 512, SMEM_SZ, stream>>>(
        Qt, Kt, Wot, bo, gout, out);
}
